// SimpleGCN_4337916969237
// MI455X (gfx1250) — matrix operand order repaired and checked by reading, not compiled, not run on hardware
//
#include <hip/hip_runtime.h>

typedef __attribute__((ext_vector_type(16))) _Float16 v16h;
typedef __attribute__((ext_vector_type(8)))  float    v8f;
typedef __attribute__((ext_vector_type(4)))  int      v4i;

#define NN   50000
#define EE   600000
#define DIN  256
#define DH   128
#define LLAY 3

#if defined(__AMDGCN__) && __has_builtin(__builtin_amdgcn_global_load_async_to_lds_b128) && __has_builtin(__builtin_amdgcn_s_wait_asynccnt)
#define HAVE_ASYNC_LDS 1
typedef __attribute__((address_space(1))) v4i* as1_v4i_p;
typedef __attribute__((address_space(3))) v4i* as3_v4i_p;
#else
#define HAVE_ASYNC_LDS 0
#endif

// Cooperatively stage `bytes` (16B-multiple) from global `src` into LDS `dst`.
// Every thread in the block must execute this; follow with __syncthreads().
__device__ inline void stage_to_lds(const char* __restrict__ src, char* dst,
                                    int bytes, int tid, int nthreads) {
#if HAVE_ASYNC_LDS
    for (int off = tid * 16; off < bytes; off += nthreads * 16) {
        __builtin_amdgcn_global_load_async_to_lds_b128(
            (as1_v4i_p)(src + off), (as3_v4i_p)(dst + off), 0, 0);
    }
    __builtin_amdgcn_s_wait_asynccnt(0);
#else
    for (int off = tid * 16; off < bytes; off += nthreads * 16)
        *(v16h*)(dst + off) = *(const v16h*)(src + off);
#endif
}

// ---------------------------------------------------------------- utilities
__global__ void k_fill(float* p, float v, long n) {
    long i = (long)blockIdx.x * blockDim.x + threadIdx.x;
    if (i < n) p[i] = v;
}

__global__ void k_deg(const int* __restrict__ ei, float* __restrict__ deg) {
    int e = blockIdx.x * blockDim.x + threadIdx.x;
    if (e < EE) atomicAdd(&deg[ei[EE + e]], 1.0f);
}

__global__ void k_dinv(const float* __restrict__ deg, float* __restrict__ dinv) {
    int i = blockIdx.x * blockDim.x + threadIdx.x;
    if (i < NN) dinv[i] = rsqrtf(deg[i]);   // deg >= 1 (self loop), so >0 always
}

// Pack a f32 [K x 128] weight matrix into WMMA B-fragment order, f16.
// Layout index: (((kt*8 + nt)*32 + lane)*16 + j)
//   column n = nt*16 + (lane&15);  k = kt*32 + (lane>>4)*8 + (j&7) + (j>>3)*16
__global__ void k_pack_b(const float* __restrict__ W, _Float16* __restrict__ out, int K) {
    int idx   = blockIdx.x * blockDim.x + threadIdx.x;
    int total = (K >> 5) * 8 * 32 * 16;
    if (idx >= total) return;
    int j    = idx & 15;
    int lane = (idx >> 4) & 31;
    int nt   = (idx >> 9) & 7;
    int kt   = idx >> 12;
    int n = nt * 16 + (lane & 15);
    int k = kt * 32 + ((lane >> 4) << 3) + (j & 7) + ((j >> 3) << 4);
    out[idx] = (_Float16)W[k * DH + n];
}

// Load an A fragment (16 rows x 32 K) for this wave from row-major f32 src.
__device__ inline v16h load_a_frag(const float* __restrict__ base_rowK, int K,
                                   int rowBase, int lane, int nrows, int kt) {
    int m    = lane & 15;
    int half = lane >> 4;
    int row  = rowBase + m;
    if (row >= nrows) row = nrows - 1;           // keep EXEC all-ones
    const float* ap = base_rowK + (size_t)row * K + kt * 32 + half * 8;
    float af[16];
    *(float4*)(af)      = *(const float4*)(ap);
    *(float4*)(af + 4)  = *(const float4*)(ap + 4);
    *(float4*)(af + 8)  = *(const float4*)(ap + 16);
    *(float4*)(af + 12) = *(const float4*)(ap + 20);
    v16h a;
#pragma unroll
    for (int j = 0; j < 16; ++j) a[j] = (_Float16)af[j];
    return a;
}

// ---------------------------------------------------------------- GEMM: out[N,128] = act(A[N,K] @ Bp + bias)
// Packed B panel staged into LDS once per block (async-to-LDS on gfx1250).
__launch_bounds__(128)
__global__ void k_gemm(const float* __restrict__ A, const _Float16* __restrict__ Bp,
                       const float* __restrict__ bias, float* __restrict__ out,
                       int nrows, int K, int do_relu) {
    extern __shared__ char smem[];
    int ktiles = K >> 5;
    int bytes  = ktiles * 8 * 32 * 32;           // packed-B panel size

    stage_to_lds((const char*)Bp, smem, bytes, threadIdx.x, blockDim.x);
    __syncthreads();

    int wave = threadIdx.x >> 5;
    int lane = threadIdx.x & 31;
    int rowBase = blockIdx.x * 64 + wave * 16;
    if (rowBase >= nrows) return;                // wave-uniform exit (after barrier)

    v8f c[8];
#pragma unroll
    for (int t = 0; t < 8; ++t)
#pragma unroll
        for (int i = 0; i < 8; ++i) c[t][i] = 0.0f;

    const v16h* bs = (const v16h*)smem;
    for (int kt = 0; kt < ktiles; ++kt) {
        v16h a = load_a_frag(A, K, rowBase, lane, nrows, kt);
        v16h bf[8];
#pragma unroll
        for (int nt = 0; nt < 8; ++nt) bf[nt] = bs[(kt * 8 + nt) * 32 + lane];
#pragma unroll
        for (int nt = 0; nt < 8; ++nt)
            c[nt] = __builtin_amdgcn_wmma_f32_16x16x32_f16(
                false, a, false, bf[nt], (short)0, c[nt], false, false);
    }

    int n0 = lane & 15, half = lane >> 4;
#pragma unroll
    for (int nt = 0; nt < 8; ++nt) {
        int col = nt * 16 + n0;
        float bv = bias ? bias[col] : 0.0f;
#pragma unroll
        for (int r = 0; r < 8; ++r) {
            int row = rowBase + half * 8 + r;
            if (row < nrows) {
                float v = c[nt][r] + bv;
                if (do_relu) v = fmaxf(v, 0.0f);
                out[(size_t)row * DH + col] = v;
            }
        }
    }
}

// ---------------------------------------------------------------- edge scatter: one wave per edge
__launch_bounds__(256)
__global__ void k_scatter(const int* __restrict__ ei, const float* __restrict__ dinv,
                          const float* __restrict__ xw, float* __restrict__ agg) {
    int wid  = (blockIdx.x * blockDim.x + threadIdx.x) >> 5;
    int lane = threadIdx.x & 31;
    if (wid >= EE) return;
    int r = ei[wid], c = ei[EE + wid];
    float nrm = dinv[r] * dinv[c];
    const float* src = xw + (size_t)r * DH;
    float*       dst = agg + (size_t)c * DH;
#pragma unroll
    for (int k = 0; k < 4; ++k) {
        int f = lane + 32 * k;
        atomicAdd(dst + f, src[f] * nrm);
    }
}

// h_out = relu(h + agg + xw*dinv^2 (self loop) + b)
__global__ void k_update(const float* __restrict__ h_in, const float* __restrict__ xw,
                         const float* __restrict__ agg, const float* __restrict__ dinv,
                         const float* __restrict__ b, float* __restrict__ h_out) {
    long idx = (long)blockIdx.x * blockDim.x + threadIdx.x;
    if (idx >= (long)NN * DH) return;
    int node = (int)(idx >> 7);
    int feat = (int)(idx & 127);
    float di = dinv[node];
    float v  = h_in[idx] + agg[idx] + xw[idx] * di * di + b[feat];
    h_out[idx] = fmaxf(v, 0.0f);
}

// ---------------------------------------------------------------- fused MLP heads
__device__ inline float sigmoidf_(float x) { return 1.0f / (1.0f + __expf(-x)); }

__launch_bounds__(128)
__global__ void k_heads(const float* __restrict__ H,
                        const _Float16* __restrict__ Bpa, const float* __restrict__ b1a,
                        const float* __restrict__ w2a, const float* __restrict__ b2a,
                        const _Float16* __restrict__ Bpb, const float* __restrict__ b1b,
                        const float* __restrict__ w2b, const float* __restrict__ b2b,
                        float* __restrict__ node_pred, float* __restrict__ origin_pred) {
    extern __shared__ char smem[];
    const int panel = 4 * 8 * 32 * 32;           // 32 KB per head matrix
    stage_to_lds((const char*)Bpa, smem, panel, threadIdx.x, blockDim.x);
    stage_to_lds((const char*)Bpb, smem + panel, panel, threadIdx.x, blockDim.x);
    __syncthreads();

    int wave = threadIdx.x >> 5;
    int lane = threadIdx.x & 31;
    int rowBase = blockIdx.x * 64 + wave * 16;
    if (rowBase >= NN) return;

    v8f ca[8], cb[8];
#pragma unroll
    for (int t = 0; t < 8; ++t)
#pragma unroll
        for (int i = 0; i < 8; ++i) { ca[t][i] = 0.0f; cb[t][i] = 0.0f; }

    const v16h* bsA = (const v16h*)smem;
    const v16h* bsB = (const v16h*)(smem + panel);
    for (int kt = 0; kt < 4; ++kt) {
        v16h a = load_a_frag(H, DH, rowBase, lane, NN, kt);
        v16h bf[8];
#pragma unroll
        for (int nt = 0; nt < 8; ++nt) bf[nt] = bsA[(kt * 8 + nt) * 32 + lane];
#pragma unroll
        for (int nt = 0; nt < 8; ++nt)
            ca[nt] = __builtin_amdgcn_wmma_f32_16x16x32_f16(
                false, a, false, bf[nt], (short)0, ca[nt], false, false);
#pragma unroll
        for (int nt = 0; nt < 8; ++nt) bf[nt] = bsB[(kt * 8 + nt) * 32 + lane];
#pragma unroll
        for (int nt = 0; nt < 8; ++nt)
            cb[nt] = __builtin_amdgcn_wmma_f32_16x16x32_f16(
                false, a, false, bf[nt], (short)0, cb[nt], false, false);
    }

    int n0 = lane & 15, half = lane >> 4;
    float pa[8], pb[8];
#pragma unroll
    for (int r = 0; r < 8; ++r) { pa[r] = 0.0f; pb[r] = 0.0f; }
#pragma unroll
    for (int nt = 0; nt < 8; ++nt) {
        int col = nt * 16 + n0;
        float ba1 = b1a[col], wa2 = w2a[col];
        float bb1 = b1b[col], wb2 = w2b[col];
#pragma unroll
        for (int r = 0; r < 8; ++r) {
            pa[r] += fmaxf(ca[nt][r] + ba1, 0.0f) * wa2;
            pb[r] += fmaxf(cb[nt][r] + bb1, 0.0f) * wb2;
        }
    }
#pragma unroll
    for (int r = 0; r < 8; ++r) {
        float sa = pa[r], sb = pb[r];
#pragma unroll
        for (int m = 1; m < 16; m <<= 1) {     // reduce across 16-lane halves
            sa += __shfl_xor(sa, m, 32);
            sb += __shfl_xor(sb, m, 32);
        }
        if (n0 == 0) {
            int row = rowBase + half * 8 + r;
            if (row < NN) {
                node_pred[row]   = sigmoidf_(sa + b2a[0]);
                origin_pred[row] = sigmoidf_(sb + b2b[0]);
            }
        }
    }
}

// ---------------------------------------------------------------- launch
extern "C" void kernel_launch(void* const* d_in, const int* in_sizes, int n_in,
                              void* d_out, int out_size, void* d_ws, size_t ws_size,
                              hipStream_t stream) {
    const float* x     = (const float*)d_in[0];
    const int*   ei    = (const int*)d_in[1];
    const float* w_in  = (const float*)d_in[2];
    const float* b_in  = (const float*)d_in[3];
    const float* w_gcn = (const float*)d_in[4];
    const float* b_gcn = (const float*)d_in[5];
    const float* w_nc1 = (const float*)d_in[6];
    const float* b_nc1 = (const float*)d_in[7];
    const float* w_nc2 = (const float*)d_in[8];
    const float* b_nc2 = (const float*)d_in[9];
    const float* w_oc1 = (const float*)d_in[10];
    const float* b_oc1 = (const float*)d_in[11];
    const float* w_oc2 = (const float*)d_in[12];
    const float* b_oc2 = (const float*)d_in[13];

    // workspace carve-up
    char* ws = (char*)d_ws;
    float* deg  = (float*)ws;               ws += (size_t)NN * 4;
    float* dinv = (float*)ws;               ws += (size_t)NN * 4;
    float* h    = (float*)ws;               ws += (size_t)NN * DH * 4;
    float* xw   = (float*)ws;               ws += (size_t)NN * DH * 4;
    float* agg  = (float*)ws;               ws += (size_t)NN * DH * 4;
    _Float16* w_in_p  = (_Float16*)ws;      ws += (size_t)(DIN / 32) * 4096 * 2;  // 8 ktiles
    _Float16* w_gcn_p = (_Float16*)ws;      ws += (size_t)LLAY * 4 * 4096 * 2;    // 4 ktiles each
    _Float16* w_nc1_p = (_Float16*)ws;      ws += (size_t)4 * 4096 * 2;
    _Float16* w_oc1_p = (_Float16*)ws;      ws += (size_t)4 * 4096 * 2;

    float* out         = (float*)d_out;
    float* node_pred   = out;
    float* origin_pred = out + NN;
    float* err_type    = out + 2 * (size_t)NN;          // N*4 zeros
    float* h_final     = out + 6 * (size_t)NN;          // N*128

    const int T = 256;
    // degrees / normalization
    k_fill<<<(NN + T - 1) / T, T, 0, stream>>>(deg, 1.0f, NN);
    k_deg<<<(EE + T - 1) / T, T, 0, stream>>>(ei, deg);
    k_dinv<<<(NN + T - 1) / T, T, 0, stream>>>(deg, dinv);

    // pack weights into WMMA fragment order (f16)
    k_pack_b<<<(8 * 4096 + T - 1) / T, T, 0, stream>>>(w_in, w_in_p, DIN);
    for (int l = 0; l < LLAY; ++l)
        k_pack_b<<<(4 * 4096 + T - 1) / T, T, 0, stream>>>(
            w_gcn + (size_t)l * DH * DH, w_gcn_p + (size_t)l * 4 * 4096, DH);
    k_pack_b<<<(4 * 4096 + T - 1) / T, T, 0, stream>>>(w_nc1, w_nc1_p, DH);
    k_pack_b<<<(4 * 4096 + T - 1) / T, T, 0, stream>>>(w_oc1, w_oc1_p, DH);

    int gemm_blocks = (NN + 63) / 64;
    // h = relu(x @ w_in + b_in)   (64 KB LDS: packed w_in panel)
    k_gemm<<<gemm_blocks, 128, 8 * 8192, stream>>>(x, w_in_p, b_in, h, NN, DIN, 1);

    long nd = (long)NN * DH;
    for (int l = 0; l < LLAY; ++l) {
        k_gemm<<<gemm_blocks, 128, 4 * 8192, stream>>>(
            h, w_gcn_p + (size_t)l * 4 * 4096, nullptr, xw, NN, DH, 0);
        k_fill<<<(int)((nd + T - 1) / T), T, 0, stream>>>(agg, 0.0f, nd);
        k_scatter<<<(EE + 7) / 8, 256, 0, stream>>>(ei, dinv, xw, agg);
        float* hdst = (l == LLAY - 1) ? h_final : h;
        k_update<<<(int)((nd + T - 1) / T), T, 0, stream>>>(
            h, xw, agg, dinv, b_gcn + (size_t)l * DH, hdst);
    }

    // heads (fused two MLPs, 64 KB LDS) + zero error_type
    k_heads<<<gemm_blocks, 128, 2 * 32768, stream>>>(h_final,
        w_nc1_p, b_nc1, w_nc2, b_nc2,
        w_oc1_p, b_oc1, w_oc2, b_oc2,
        node_pred, origin_pred);
    k_fill<<<(int)(((long)NN * 4 + T - 1) / T), T, 0, stream>>>(err_type, 0.0f, (long)NN * 4);
}
